// SlotAttentionWithPositions_79250736546364
// MI455X (gfx1250) — compile-verified, matrix operand order checked by reading, not run on hardware
//
#include <hip/hip_runtime.h>
#include <hip/hip_bf16.h>
#include <math.h>

// ---------------------------------------------------------------------------
// SlotAttentionWithPositions for MI455X (gfx1250, wave32, WMMA).
// bf16 WMMA (f32 accum) for all large GEMMs; async-to-LDS A-tile staging;
// norm_input LN fused into the fm2 GEMM epilogue (saves ~268MB HBM traffic).
// ---------------------------------------------------------------------------

typedef __bf16 bf16_t;
typedef bf16_t v16bf __attribute__((ext_vector_type(16)));
typedef float  v8f   __attribute__((ext_vector_type(8)));

#define B_     32
#define D_     256
#define RES_   64
#define NTOK_  4096
#define NSLOT_ 8
#define SPAD_  16      // slots padded to one WMMA M-tile
#define NITER_ 3
#define SCALE_ 0.0625f // D^-0.5
#define EPS_   1e-8f
#define LNEPS_ 1e-5f

// ---- WMMA fragment helpers -------------------------------------------------
// 16-bit A fragment layout (ISA 7.12.2): lane<16 -> K {0..7, 16..23},
// lane>=16 -> K {8..15, 24..31}; i.e. two contiguous 16B runs per lane.
static __device__ __forceinline__ v16bf load_frag(const bf16_t* __restrict__ base,
                                                  int ld, int k0) {
  const int lane = threadIdx.x & 31;
  const int row  = lane & 15;
  const int kb   = (lane >> 4) * 8;
  const bf16_t* p = base + (size_t)row * ld + (k0 + kb);
  union { v16bf v; uint4 u[2]; } f;
  f.u[0] = *(const uint4*)(p);
  f.u[1] = *(const uint4*)(p + 16);
  return f.v;
}

// same pattern but from LDS (pointer derived from __shared__ -> ds_load_b128)
static __device__ __forceinline__ v16bf load_frag_s(const bf16_t* base,
                                                    int ld, int k0) {
  const int lane = threadIdx.x & 31;
  const int row  = lane & 15;
  const int kb   = (lane >> 4) * 8;
  const bf16_t* p = base + row * ld + (k0 + kb);
  union { v16bf v; uint4 u[2]; } f;
  f.u[0] = *(const uint4*)(p);
  f.u[1] = *(const uint4*)(p + 16);
  return f.v;
}

static __device__ __forceinline__ v8f zero_acc() {
  v8f a;
#pragma unroll
  for (int i = 0; i < 8; ++i) a[i] = 0.0f;
  return a;
}

// ---- block reduction (blockDim.x == 256) -----------------------------------
static __device__ __forceinline__ float block_sum(float v, float* buf) {
  const int tid = threadIdx.x;
  buf[tid] = v;
  __syncthreads();
#pragma unroll
  for (int off = 128; off > 0; off >>= 1) {
    if (tid < off) buf[tid] += buf[tid + off];
    __syncthreads();
  }
  float r = buf[0];
  __syncthreads();
  return r;
}

// ---------------------------------------------------------------------------
// 0) zero the global-LN stats
__global__ void zero_stats_kernel(float* stats) {
  if (threadIdx.x < 2 * B_) stats[threadIdx.x] = 0.0f;
}

// 1) weight prep: f32 [K,N] -> bf16 transposed [N,K]
__global__ __launch_bounds__(256) void prep_wt_kernel(const float* __restrict__ W,
                                                      bf16_t* __restrict__ Wt) {
  int idx = blockIdx.x * 256 + threadIdx.x;   // grid: 256 blocks
  int n = idx >> 8, k = idx & 255;
  Wt[n * D_ + k] = (bf16_t)W[k * D_ + n];
}

// 2) slots init: slots = loc + exp(logscale) * eps   (slots live in d_out)
__global__ __launch_bounds__(256) void init_slots_kernel(const float* __restrict__ loc,
                                                         const float* __restrict__ logscale,
                                                         const float* __restrict__ eps,
                                                         float* __restrict__ slots) {
  int d = threadIdx.x;
  size_t row = blockIdx.x;                    // grid: B*NSLOT
  slots[row * D_ + d] = loc[d] + __expf(logscale[d]) * eps[row * D_ + d];
}

// 3) pos-embed + NCHW->NTD transpose (LDS tile) + global LN partial stats
__global__ __launch_bounds__(256) void posembed_kernel(const float* __restrict__ x,
                                                       const float* __restrict__ pos_w,
                                                       const float* __restrict__ pos_b,
                                                       float* __restrict__ tokf,
                                                       float* __restrict__ stats) {
  __shared__ float tile[64][65];
  __shared__ float red[256];
  const int b  = blockIdx.y;
  const int t0 = (blockIdx.x & 63) * 64;      // NTOK/64 = 64 tiles
  const int d0 = (blockIdx.x >> 6) * 64;      // D/64   = 4 tiles
  const int tid = threadIdx.x;
  const int cx = tid & 63, ry = tid >> 6;     // 64-wide coalesced, 4 rows/pass
  for (int dy = ry; dy < 64; dy += 4)
    tile[dy][cx] = x[((size_t)(b * D_ + d0 + dy)) * NTOK_ + t0 + cx];
  __syncthreads();
  const int d = d0 + cx;
  const float w0 = pos_w[0 * D_ + d], w1 = pos_w[1 * D_ + d];
  const float w2 = pos_w[2 * D_ + d], w3 = pos_w[3 * D_ + d];
  const float pb = pos_b[d];
  float lsum = 0.0f, lsq = 0.0f;
  for (int ty = ry; ty < 64; ty += 4) {
    int t = t0 + ty;
    float xx = (float)(t >> 6) * (1.0f / 63.0f);
    float yy = (float)(t & 63) * (1.0f / 63.0f);
    float p = xx * w0 + yy * w1 + (1.0f - xx) * w2 + (1.0f - yy) * w3 + pb;
    float h = tile[cx][ty] + p;
    tokf[((size_t)b * NTOK_ + t) * D_ + d] = h;
    lsum += h; lsq += h * h;
  }
  float s1 = block_sum(lsum, red);
  float s2 = block_sum(lsq, red);
  if (tid == 0) {
    atomicAdd(&stats[b * 2 + 0], s1);
    atomicAdd(&stats[b * 2 + 1], s2);
  }
}

// 4) joint LN over [NTOK,D] + enc_g/enc_b, cast to bf16
__global__ __launch_bounds__(256) void encnorm_kernel(const float* __restrict__ tokf,
                                                      const float* __restrict__ stats,
                                                      const float* __restrict__ enc_g,
                                                      const float* __restrict__ enc_b,
                                                      bf16_t* __restrict__ out) {
  const int b = blockIdx.x >> 12;             // grid: B*NTOK
  const int t = blockIdx.x & 4095;
  const int d = threadIdx.x;
  const float cnt = (float)(NTOK_ * D_);
  float m = stats[b * 2 + 0] / cnt;
  float var = stats[b * 2 + 1] / cnt - m * m;
  float rstd = rsqrtf(var + LNEPS_);
  size_t idx = ((size_t)b * NTOK_ + t) * D_ + d;
  float v = (tokf[idx] - m) * rstd * enc_g[t * D_ + d] + enc_b[t * D_ + d];
  out[idx] = (bf16_t)v;
}

// 5) WMMA GEMM: C[M,256] = A[M,256](bf16) x Bt[256,256](bf16,[N][K]) + bias.
//    One workgroup = 32 rows x full 256 cols (8 waves: 2 M-tiles x 4 N-groups).
//    A block (16KB) staged to LDS with global_load_async_to_lds_b128.
//    mode 0: relu -> bf16
//    mode 1: relu -> per-row LayerNorm(lng,lnb) -> bf16   (norm_input fused)
//    mode 2: -> bf16
//    mode 3: -> bf16 transposed vT[b][d][t] (packed b128 stores)
__global__ __launch_bounds__(256) void gemm_bf16_kernel(const bf16_t* __restrict__ A,
                                                        const bf16_t* __restrict__ Bt,
                                                        const float* __restrict__ bias,
                                                        const float* __restrict__ lng,
                                                        const float* __restrict__ lnb,
                                                        bf16_t* __restrict__ outb,
                                                        int M, int mode) {
  __shared__ bf16_t sA[32 * D_];              // 16 KB
  __shared__ float  sC[32][264];              // 33 KB (mode 1 only)
  const int tid = threadIdx.x;
  const int m0 = blockIdx.x * 32;             // grid: M/32 blocks
  // ---- async stage A[32][256] bf16 -> LDS (4 x b128 per thread) ----
  {
    const bf16_t* Ablk = A + (size_t)m0 * D_;
    uint32_t lbase = (uint32_t)(uintptr_t)(&sA[0]);
#pragma unroll
    for (int i = 0; i < 4; ++i) {
      int chunk = i * 256 + tid;              // 16-byte chunks, 1024 total
      uint64_t ga = (uint64_t)(uintptr_t)(Ablk + chunk * 8);
      uint32_t la = lbase + chunk * 16;
      asm volatile("global_load_async_to_lds_b128 %0, %1, off"
                   :: "v"(la), "v"(ga) : "memory");
    }
    asm volatile("s_wait_asynccnt 0" ::: "memory");
  }
  __syncthreads();

  const int w   = tid >> 5;                   // wave 0..7
  const int mtl = w >> 2;                     // local M tile 0..1
  const int ng  = w & 3;                      // N group 0..3 (64 cols)
  const bf16_t* Als = sA + mtl * 16 * D_;
  v8f acc[4];
#pragma unroll
  for (int j = 0; j < 4; ++j) acc[j] = zero_acc();
#pragma unroll
  for (int k0 = 0; k0 < D_; k0 += 32) {
    v16bf fa = load_frag_s(Als, D_, k0);
#pragma unroll
    for (int j = 0; j < 4; ++j) {
      v16bf fb = load_frag(Bt + (size_t)(ng * 64 + j * 16) * D_, D_, k0);
      acc[j] = __builtin_amdgcn_wmma_f32_16x16x32_bf16(
          false, fa, false, fb, (short)0, acc[j], false, false);
    }
  }
  const int lane = tid & 31;
  const int n = lane & 15, mh = (lane >> 4) * 8;

  if (mode == 1) {
    // park f32 in LDS, per-row LN, emit bf16
#pragma unroll
    for (int j = 0; j < 4; ++j) {
      int ncol = ng * 64 + j * 16 + n;
      float bv = bias[ncol];
#pragma unroll
      for (int r = 0; r < 8; ++r)
        sC[mtl * 16 + mh + r][ncol] = fmaxf(acc[j][r] + bv, 0.0f);
    }
    __syncthreads();
    // wave w handles rows w*4 .. w*4+3 (32 lanes x 8 cols each)
    for (int rr = 0; rr < 4; ++rr) {
      int row = w * 4 + rr;
      float s = 0.0f, sq = 0.0f;
      for (int c = lane; c < D_; c += 32) {
        float vv = sC[row][c];
        s += vv; sq += vv * vv;
      }
#pragma unroll
      for (int off = 16; off > 0; off >>= 1) {
        s  += __shfl_xor(s, off, 32);
        sq += __shfl_xor(sq, off, 32);
      }
      float m = s * (1.0f / D_);
      float var = sq * (1.0f / D_) - m * m;
      float rstd = rsqrtf(var + LNEPS_);
      size_t grow = (size_t)m0 + row;
      for (int c = lane; c < D_; c += 32)
        outb[grow * D_ + c] = (bf16_t)((sC[row][c] - m) * rstd * lng[c] + lnb[c]);
    }
    return;
  }

#pragma unroll
  for (int j = 0; j < 4; ++j) {
    int ncol = ng * 64 + j * 16 + n;
    float bv = bias[ncol];
    if (mode == 3) {
      // vT[b][d][t]: rows in a lane are consecutive t -> pack one b128 store
      size_t mb = (size_t)m0 + mtl * 16 + mh;
      size_t bb = mb >> 12, tb = mb & 4095;
      union { uint4 u; bf16_t h[8]; } pk;
#pragma unroll
      for (int r = 0; r < 8; ++r) pk.h[r] = (bf16_t)(acc[j][r] + bv);
      *(uint4*)&outb[(bb * D_ + ncol) * NTOK_ + tb] = pk.u;
    } else {
#pragma unroll
      for (int r = 0; r < 8; ++r) {
        float v = acc[j][r] + bv;
        if (mode == 0) v = fmaxf(v, 0.0f);
        size_t m = (size_t)m0 + mtl * 16 + mh + r;
        outb[m * D_ + ncol] = (bf16_t)v;
      }
    }
  }
}

// 7) slot LN + q projection (+ rowsum reset); writes bf16 qpad[B][16][D]
__global__ __launch_bounds__(256) void q_kernel(const float* __restrict__ slots,
                                                const float* __restrict__ slot_g,
                                                const float* __restrict__ slot_b,
                                                const float* __restrict__ q_w,
                                                const float* __restrict__ q_b,
                                                bf16_t* __restrict__ qpad,
                                                float* __restrict__ rowsum) {
  __shared__ float red[256];
  __shared__ float sln[256];
  const int b = blockIdx.x >> 4;              // grid: B*SPAD
  const int s = blockIdx.x & 15;
  const int d = threadIdx.x;
  if (d == 0) rowsum[b * SPAD_ + s] = 0.0f;
  size_t qi = ((size_t)b * SPAD_ + s) * D_ + d;
  if (s >= NSLOT_) { qpad[qi] = (bf16_t)0.0f; return; }
  float v = slots[((size_t)b * NSLOT_ + s) * D_ + d];
  float m = block_sum(v, red) * (1.0f / D_);
  float dv = v - m;
  float var = block_sum(dv * dv, red) * (1.0f / D_);
  float rstd = rsqrtf(var + LNEPS_);
  sln[d] = dv * rstd * slot_g[d] + slot_b[d];
  __syncthreads();
  float q = q_b[d];
  for (int k = 0; k < D_; ++k) q += sln[k] * q_w[k * D_ + d];
  qpad[qi] = (bf16_t)q;
}

// 8) dots = (qpad @ k^T) * SCALE, WMMA, per-batch. grid: B*16 chunks, 8 waves
__global__ __launch_bounds__(256) void dots_kernel(const bf16_t* __restrict__ qpad,
                                                   const bf16_t* __restrict__ kmat,
                                                   float* __restrict__ dots) {
  const int b = blockIdx.x >> 4;
  const int chunk = blockIdx.x & 15;
  const int w = threadIdx.x >> 5;
  const int n0 = chunk * 256 + w * 32;        // token offset
  const bf16_t* A = qpad + (size_t)b * SPAD_ * D_;
  v8f acc0 = zero_acc(), acc1 = zero_acc();
#pragma unroll
  for (int k0 = 0; k0 < D_; k0 += 32) {
    v16bf fa = load_frag(A, D_, k0);
    v16bf fb0 = load_frag(kmat + ((size_t)b * NTOK_ + n0) * D_, D_, k0);
    v16bf fb1 = load_frag(kmat + ((size_t)b * NTOK_ + n0 + 16) * D_, D_, k0);
    acc0 = __builtin_amdgcn_wmma_f32_16x16x32_bf16(false, fa, false, fb0, (short)0, acc0, false, false);
    acc1 = __builtin_amdgcn_wmma_f32_16x16x32_bf16(false, fa, false, fb1, (short)0, acc1, false, false);
  }
  const int lane = threadIdx.x & 31;
  const int n = lane & 15, mh = (lane >> 4) * 8;
#pragma unroll
  for (int r = 0; r < 8; ++r) {
    size_t srow = (size_t)b * SPAD_ + mh + r;
    dots[srow * NTOK_ + n0 + n]      = acc0[r] * SCALE_;
    dots[srow * NTOK_ + n0 + 16 + n] = acc1[r] * SCALE_;
  }
}

// 9) softmax over SLOT axis + EPS; write bf16 attn (padded rows zeroed) and
//    accumulate per-(b,s) token sums for renormalization.
__global__ __launch_bounds__(256) void softmax_kernel(const float* __restrict__ dots,
                                                      bf16_t* __restrict__ attnb,
                                                      float* __restrict__ rowsum) {
  __shared__ float red[256];
  const int b = blockIdx.x >> 4;              // grid: B*16
  const int t = (blockIdx.x & 15) * 256 + threadIdx.x;
  float v[NSLOT_];
  float vmax = -3.0e38f;
#pragma unroll
  for (int s = 0; s < NSLOT_; ++s) {
    v[s] = dots[((size_t)b * SPAD_ + s) * NTOK_ + t];
    vmax = fmaxf(vmax, v[s]);
  }
  float sum = 0.0f;
#pragma unroll
  for (int s = 0; s < NSLOT_; ++s) { v[s] = __expf(v[s] - vmax); sum += v[s]; }
  float inv = 1.0f / sum;
  float part[NSLOT_];
#pragma unroll
  for (int s = 0; s < NSLOT_; ++s) {
    float a = v[s] * inv + EPS_;
    part[s] = a;
    attnb[((size_t)b * SPAD_ + s) * NTOK_ + t] = (bf16_t)a;
  }
#pragma unroll
  for (int s = NSLOT_; s < SPAD_; ++s)
    attnb[((size_t)b * SPAD_ + s) * NTOK_ + t] = (bf16_t)0.0f;
  for (int s = 0; s < NSLOT_; ++s) {
    float ps = block_sum(part[s], red);
    if (threadIdx.x == 0) atomicAdd(&rowsum[b * SPAD_ + s], ps);
  }
}

// 10) upd = (attn @ v) / rowsum, WMMA over K=NTOK. grid: B blocks, 8 waves
__global__ __launch_bounds__(256) void upd_kernel(const bf16_t* __restrict__ attnb,
                                                  const bf16_t* __restrict__ vt,
                                                  const float* __restrict__ rowsum,
                                                  float* __restrict__ updp) {
  const int b = blockIdx.x;
  const int w = threadIdx.x >> 5;
  const int n0 = w * 32;                      // output channel offset
  const bf16_t* A = attnb + (size_t)b * SPAD_ * NTOK_;
  v8f acc0 = zero_acc(), acc1 = zero_acc();
  for (int k0 = 0; k0 < NTOK_; k0 += 32) {
    v16bf fa = load_frag(A, NTOK_, k0);
    v16bf fb0 = load_frag(vt + ((size_t)b * D_ + n0) * NTOK_, NTOK_, k0);
    v16bf fb1 = load_frag(vt + ((size_t)b * D_ + n0 + 16) * NTOK_, NTOK_, k0);
    acc0 = __builtin_amdgcn_wmma_f32_16x16x32_bf16(false, fa, false, fb0, (short)0, acc0, false, false);
    acc1 = __builtin_amdgcn_wmma_f32_16x16x32_bf16(false, fa, false, fb1, (short)0, acc1, false, false);
  }
  const int lane = threadIdx.x & 31;
  const int n = lane & 15, mh = (lane >> 4) * 8;
#pragma unroll
  for (int r = 0; r < 8; ++r) {
    int s = mh + r;
    if (s < NSLOT_) {
      float inv = 1.0f / rowsum[b * SPAD_ + s];
      size_t o = ((size_t)b * SPAD_ + s) * D_;
      updp[o + n0 + n]      = acc0[r] * inv;
      updp[o + n0 + 16 + n] = acc1[r] * inv;
    }
  }
}

// 11) GRU cell + pre-LN + residual MLP, one block per (b,s)
__global__ __launch_bounds__(256) void gru_kernel(const float* __restrict__ updp,
                                                  float* __restrict__ slots,
                                                  const float* __restrict__ wih,
                                                  const float* __restrict__ whh,
                                                  const float* __restrict__ bih,
                                                  const float* __restrict__ bhh,
                                                  const float* __restrict__ pre_g,
                                                  const float* __restrict__ pre_b,
                                                  const float* __restrict__ st_w1,
                                                  const float* __restrict__ st_b1,
                                                  const float* __restrict__ st_w2,
                                                  const float* __restrict__ st_b2) {
  __shared__ float red[256];
  __shared__ float su[256], sp[256], sh2[256];
  const int b = blockIdx.x >> 3;              // grid: B*NSLOT
  const int s = blockIdx.x & 7;
  const int d = threadIdx.x;
  su[d] = updp[((size_t)b * SPAD_ + s) * D_ + d];
  float prev = slots[((size_t)b * NSLOT_ + s) * D_ + d];
  sp[d] = prev;
  __syncthreads();
  float ir = bih[d], iz = bih[D_ + d], in = bih[2 * D_ + d];
  float hr = bhh[d], hz = bhh[D_ + d], hn = bhh[2 * D_ + d];
  for (int k = 0; k < D_; ++k) {
    float u = su[k], p = sp[k];
    const float* wi = wih + (size_t)k * 3 * D_;
    const float* wh = whh + (size_t)k * 3 * D_;
    ir += u * wi[d]; iz += u * wi[D_ + d]; in += u * wi[2 * D_ + d];
    hr += p * wh[d]; hz += p * wh[D_ + d]; hn += p * wh[2 * D_ + d];
  }
  float r = 1.0f / (1.0f + __expf(-(ir + hr)));
  float z = 1.0f / (1.0f + __expf(-(iz + hz)));
  float nn = tanhf(in + r * hn);
  float news = (1.0f - z) * nn + z * prev;
  float m = block_sum(news, red) * (1.0f / D_);
  float dv = news - m;
  float var = block_sum(dv * dv, red) * (1.0f / D_);
  float rstd = rsqrtf(var + LNEPS_);
  sh2[d] = dv * rstd * pre_g[d] + pre_b[d];
  __syncthreads();
  float f1 = st_b1[d];
  for (int k = 0; k < D_; ++k) f1 += sh2[k] * st_w1[k * D_ + d];
  f1 = fmaxf(f1, 0.0f);
  __syncthreads();
  sh2[d] = f1;
  __syncthreads();
  float f2 = st_b2[d];
  for (int k = 0; k < D_; ++k) f2 += sh2[k] * st_w2[k * D_ + d];
  f2 = fmaxf(f2, 0.0f);
  slots[((size_t)b * NSLOT_ + s) * D_ + d] = news + f2;
}

// ---------------------------------------------------------------------------
extern "C" void kernel_launch(void* const* d_in, const int* in_sizes, int n_in,
                              void* d_out, int out_size, void* d_ws, size_t ws_size,
                              hipStream_t stream) {
  (void)in_sizes; (void)n_in; (void)out_size; (void)ws_size;
  const float* x        = (const float*)d_in[0];
  const float* eps      = (const float*)d_in[1];
  const float* pos_w    = (const float*)d_in[2];
  const float* pos_b    = (const float*)d_in[3];
  const float* enc_g    = (const float*)d_in[4];
  const float* enc_b    = (const float*)d_in[5];
  const float* fm_w1    = (const float*)d_in[6];
  const float* fm_b1    = (const float*)d_in[7];
  const float* fm_w2    = (const float*)d_in[8];
  const float* fm_b2    = (const float*)d_in[9];
  const float* in_g     = (const float*)d_in[10];
  const float* in_b     = (const float*)d_in[11];
  const float* q_w      = (const float*)d_in[12];
  const float* q_b      = (const float*)d_in[13];
  const float* k_w      = (const float*)d_in[14];
  const float* k_b      = (const float*)d_in[15];
  const float* v_w      = (const float*)d_in[16];
  const float* v_b      = (const float*)d_in[17];
  const float* gru_wih  = (const float*)d_in[18];
  const float* gru_whh  = (const float*)d_in[19];
  const float* gru_bih  = (const float*)d_in[20];
  const float* gru_bhh  = (const float*)d_in[21];
  const float* pre_g    = (const float*)d_in[22];
  const float* pre_b    = (const float*)d_in[23];
  const float* st_w1    = (const float*)d_in[24];
  const float* st_b1    = (const float*)d_in[25];
  const float* st_w2    = (const float*)d_in[26];
  const float* st_b2    = (const float*)d_in[27];
  const float* slot_g   = (const float*)d_in[28];
  const float* slot_b   = (const float*)d_in[29];
  const float* sl_loc   = (const float*)d_in[30];
  const float* sl_lsc   = (const float*)d_in[31];

  float* slots = (float*)d_out;               // [B,8,256] f32 — lives in d_out

  // workspace carve (all sections 256B-aligned)
  const size_t MTOT = (size_t)B_ * NTOK_;     // 131072
  char* w = (char*)d_ws;
  float*  tokf  = (float*)w;  w += MTOT * D_ * 4;         // 134 MB
  bf16_t* tokA  = (bf16_t*)w; w += MTOT * D_ * 2;         //  67 MB
  bf16_t* tokB  = (bf16_t*)w; w += MTOT * D_ * 2;         //  67 MB
  bf16_t* vt    = (bf16_t*)w; w += MTOT * D_ * 2;         //  67 MB  v^T [B][D][NTOK]
  bf16_t* qpad  = (bf16_t*)w; w += (size_t)B_ * SPAD_ * D_ * 2;
  bf16_t* attnb = (bf16_t*)w; w += (size_t)B_ * SPAD_ * NTOK_ * 2;
  float*  dots  = (float*)w;  w += (size_t)B_ * SPAD_ * NTOK_ * 4;
  float*  updp  = (float*)w;  w += (size_t)B_ * SPAD_ * D_ * 4;
  float*  rowsum= (float*)w;  w += (size_t)B_ * SPAD_ * 4;
  float*  stats = (float*)w;  w += 256;
  bf16_t* w1t   = (bf16_t*)w; w += (size_t)D_ * D_ * 2;
  bf16_t* w2t   = (bf16_t*)w; w += (size_t)D_ * D_ * 2;
  bf16_t* kwt   = (bf16_t*)w; w += (size_t)D_ * D_ * 2;
  bf16_t* vwt   = (bf16_t*)w; w += (size_t)D_ * D_ * 2;

  // ---- setup ----
  zero_stats_kernel<<<1, 64, 0, stream>>>(stats);
  prep_wt_kernel<<<256, 256, 0, stream>>>(fm_w1, w1t);
  prep_wt_kernel<<<256, 256, 0, stream>>>(fm_w2, w2t);
  prep_wt_kernel<<<256, 256, 0, stream>>>(k_w, kwt);
  prep_wt_kernel<<<256, 256, 0, stream>>>(v_w, vwt);
  init_slots_kernel<<<B_ * NSLOT_, 256, 0, stream>>>(sl_loc, sl_lsc, eps, slots);

  // ---- encoder ----
  posembed_kernel<<<dim3(256, B_), 256, 0, stream>>>(x, pos_w, pos_b, tokf, stats);
  encnorm_kernel<<<B_ * NTOK_, 256, 0, stream>>>(tokf, stats, enc_g, enc_b, tokA);

  const int gemm_blocks = (int)(MTOT / 32);   // one block = 32 rows x 256 cols
  gemm_bf16_kernel<<<gemm_blocks, 256, 0, stream>>>(tokA, w1t, fm_b1, nullptr, nullptr,
                                                    tokB, (int)MTOT, 0);          // relu
  gemm_bf16_kernel<<<gemm_blocks, 256, 0, stream>>>(tokB, w2t, fm_b2, in_g, in_b,
                                                    tokA, (int)MTOT, 1);          // relu+LN
  gemm_bf16_kernel<<<gemm_blocks, 256, 0, stream>>>(tokA, kwt, k_b, nullptr, nullptr,
                                                    tokB, (int)MTOT, 2);          // k
  gemm_bf16_kernel<<<gemm_blocks, 256, 0, stream>>>(tokA, vwt, v_b, nullptr, nullptr,
                                                    vt, (int)MTOT, 3);            // v^T

  // ---- slot attention iterations ----
  for (int it = 0; it < NITER_; ++it) {
    q_kernel<<<B_ * SPAD_, 256, 0, stream>>>(slots, slot_g, slot_b, q_w, q_b, qpad, rowsum);
    dots_kernel<<<B_ * 16, 256, 0, stream>>>(qpad, tokB, dots);
    softmax_kernel<<<B_ * 16, 256, 0, stream>>>(dots, attnb, rowsum);
    upd_kernel<<<B_, 256, 0, stream>>>(attnb, vt, rowsum, updp);
    gru_kernel<<<B_ * NSLOT_, 256, 0, stream>>>(updp, slots, gru_wih, gru_whh,
                                                gru_bih, gru_bhh, pre_g, pre_b,
                                                st_w1, st_b1, st_w2, st_b2);
  }
}
